// MaskAttention_30846455119890
// MI455X (gfx1250) — matrix operand order repaired and checked by reading, not compiled, not run on hardware
//
#include <hip/hip_runtime.h>
#include <hip/hip_bf16.h>

// ---------------------------------------------------------------------------
// Problem constants (from reference): B=4, Q=1024, S=4096, D=256, H=8, HD=32
// ---------------------------------------------------------------------------
#define B_  4
#define Q_  1024
#define S_  4096
#define D_  256
#define H_  8
#define HD_ 32
#define NEGV (-1000000000.0f)

typedef __attribute__((ext_vector_type(16))) _Float16 v16h;
typedef __attribute__((ext_vector_type(8)))  float    v8f;
typedef int v4i __attribute__((__vector_size__(16)));   // matches builtin param type

union F16Frag { v16h v; _Float16 h[16]; uint4 q[2]; };
union F16x8   { uint4 u; _Float16 h[8]; };

// ---------------------------------------------------------------------------
// CDNA5 async global->LDS staging (ASYNCcnt-tracked, no data VGPRs).
// Signature (from clang diagnostic): (v4i AS1* gsrc, v4i AS3* ldst, imm, imm)
// ---------------------------------------------------------------------------
#define GLOBAL_AS __attribute__((address_space(1)))
#define LDS_AS    __attribute__((address_space(3)))

#if __has_builtin(__builtin_amdgcn_global_load_async_to_lds_b128)
#define ASYNC_LDS 1
#else
#define ASYNC_LDS 0
#endif

#if ASYNC_LDS
#define STAGE16(gp, lp)                                   \
  __builtin_amdgcn_global_load_async_to_lds_b128(         \
      (GLOBAL_AS v4i*)(gp), (LDS_AS v4i*)(lp), 0, 0)
__device__ __forceinline__ void async_wait_all() {
#if __has_builtin(__builtin_amdgcn_s_wait_asynccnt)
  __builtin_amdgcn_s_wait_asynccnt(0);
#else
  asm volatile("s_wait_asynccnt 0" ::: "memory");
#endif
}
#endif

// ---------------------------------------------------------------------------
// LayerNorm over last dim (256) -> f16.  One wave per row, 8 rows/block.
// ---------------------------------------------------------------------------
__global__ __launch_bounds__(256)
void ln_kernel(const float* __restrict__ x, const float* __restrict__ g,
               const float* __restrict__ bta, _Float16* __restrict__ out) {
  int wave = threadIdx.x >> 5, lane = threadIdx.x & 31;
  int row  = blockIdx.x * 8 + wave;
  const float* xr = x + (size_t)row * D_;
  float v[8]; float s = 0.f, sq = 0.f;
#pragma unroll
  for (int k = 0; k < 8; ++k) {
    v[k] = xr[lane + k * 32];
    s += v[k]; sq += v[k] * v[k];
  }
#pragma unroll
  for (int m = 1; m < 32; m <<= 1) {
    s  += __shfl_xor(s,  m, 32);
    sq += __shfl_xor(sq, m, 32);
  }
  float mean = s * (1.0f / D_);
  float var  = sq * (1.0f / D_) - mean * mean;
  float rstd = rsqrtf(var + 1e-5f);
  _Float16* o = out + (size_t)row * D_;
#pragma unroll
  for (int k = 0; k < 8; ++k) {
    int c = lane + k * 32;
    o[c] = (_Float16)((v[k] - mean) * rstd * g[c] + bta[c]);
  }
}

// ---------------------------------------------------------------------------
// Elementwise f32 -> f16 cast.
// ---------------------------------------------------------------------------
__global__ __launch_bounds__(256)
void cast_kernel(const float* __restrict__ src, _Float16* __restrict__ dst, int n) {
  int i = blockIdx.x * 256 + threadIdx.x;
  if (i < n) dst[i] = (_Float16)src[i];
}

// ---------------------------------------------------------------------------
// GEMM: out[M,256] = A[M,256] f16 @ W[256,256]^T f16 (+bias, +optional
// residual with f32 out).  128 threads / 4 waves, 64x64 block tile,
// double-buffered LDS with async global->LDS staging pipelined under WMMAs.
// ---------------------------------------------------------------------------
__global__ __launch_bounds__(128)
void gemm256_kernel(const _Float16* __restrict__ A, const _Float16* __restrict__ W,
                    const float* __restrict__ bias, _Float16* __restrict__ outH,
                    float* __restrict__ outF, const float* __restrict__ resid) {
  __shared__ __align__(16) _Float16 lA[2][64 * 32];
  __shared__ __align__(16) _Float16 lB[2][64 * 32];
  const int t    = threadIdx.x;
  const int wave = t >> 5, lane = t & 31;
  const int m0   = blockIdx.x * 64, n0 = blockIdx.y * 64;
  const int r    = lane & 15, hg = lane >> 4;

  const v8f vzero = {0.f, 0.f, 0.f, 0.f, 0.f, 0.f, 0.f, 0.f};
  v8f acc[4];
#pragma unroll
  for (int c = 0; c < 4; ++c) acc[c] = vzero;

  const int ldr = t >> 1;          // LDS tile row staged by this thread
  const int seg = t & 1;           // 32B segment within the row
  const _Float16* gA = A + (size_t)(m0 + ldr) * D_ + seg * 16;
  const _Float16* gB = W + (size_t)(n0 + ldr) * D_ + seg * 16;
  const int loff = ldr * 32 + seg * 16;

  // ---- prologue: stage K-step 0 into buffer 0
#if ASYNC_LDS
  STAGE16(gA,     &lA[0][loff]);
  STAGE16(gA + 8, &lA[0][loff + 8]);
  STAGE16(gB,     &lB[0][loff]);
  STAGE16(gB + 8, &lB[0][loff + 8]);
  async_wait_all();
#else
  {
    const uint4* sa = reinterpret_cast<const uint4*>(gA);
    const uint4* sb = reinterpret_cast<const uint4*>(gB);
    uint4 a0 = sa[0], a1 = sa[1], b0 = sb[0], b1 = sb[1];
    *reinterpret_cast<uint4*>(&lA[0][loff])     = a0;
    *reinterpret_cast<uint4*>(&lA[0][loff + 8]) = a1;
    *reinterpret_cast<uint4*>(&lB[0][loff])     = b0;
    *reinterpret_cast<uint4*>(&lB[0][loff + 8]) = b1;
  }
#endif
  __syncthreads();

  for (int kk = 0; kk < D_ / 32; ++kk) {
    const int cur = kk & 1, nxt = cur ^ 1;
    const bool more = (kk + 1 < D_ / 32);

    // ---- stage next K-step into the other buffer (no wait yet)
#if ASYNC_LDS
    if (more) {
      const _Float16* nA = gA + (kk + 1) * 32;
      const _Float16* nB = gB + (kk + 1) * 32;
      STAGE16(nA,     &lA[nxt][loff]);
      STAGE16(nA + 8, &lA[nxt][loff + 8]);
      STAGE16(nB,     &lB[nxt][loff]);
      STAGE16(nB + 8, &lB[nxt][loff + 8]);
    }
#else
    uint4 a0, a1, b0, b1;
    if (more) {
      const uint4* sa = reinterpret_cast<const uint4*>(gA + (kk + 1) * 32);
      const uint4* sb = reinterpret_cast<const uint4*>(gB + (kk + 1) * 32);
      a0 = sa[0]; a1 = sa[1]; b0 = sb[0]; b1 = sb[1];
    }
#endif

    // ---- compute on current buffer
    // A fragment: elems 0-7 -> k=hg*8+i, 8-15 -> k=16+hg*8+(i-8)
    F16Frag af;
    const _Float16* ab = &lA[cur][(wave * 16 + r) * 32 + hg * 8];
    af.q[0] = *reinterpret_cast<const uint4*>(ab);
    af.q[1] = *reinterpret_cast<const uint4*>(ab + 16);
#pragma unroll
    for (int c = 0; c < 4; ++c) {
      // B fragment: lane column n = c*16 + r, elems 0-7 -> k=hg*8+i, 8-15 -> k=16+hg*8+(i-8)
      F16Frag bf;
      const _Float16* bb = &lB[cur][(c * 16 + r) * 32 + hg * 8];
      bf.q[0] = *reinterpret_cast<const uint4*>(bb);
      bf.q[1] = *reinterpret_cast<const uint4*>(bb + 16);
      acc[c] = __builtin_amdgcn_wmma_f32_16x16x32_f16(
          false, af.v, false, bf.v, (short)0, acc[c], false, false);
    }

    // ---- finish staging, publish buffers
#if ASYNC_LDS
    if (more) async_wait_all();
#else
    if (more) {
      *reinterpret_cast<uint4*>(&lA[nxt][loff])     = a0;
      *reinterpret_cast<uint4*>(&lA[nxt][loff + 8]) = a1;
      *reinterpret_cast<uint4*>(&lB[nxt][loff])     = b0;
      *reinterpret_cast<uint4*>(&lB[nxt][loff + 8]) = b1;
    }
#endif
    __syncthreads();
  }

#pragma unroll
  for (int c = 0; c < 4; ++c) {
#pragma unroll
    for (int j = 0; j < 8; ++j) {
      int m = m0 + wave * 16 + j + 8 * hg;
      int n = n0 + c * 16 + r;
      float val = acc[c][j] + bias[n];
      if (outF) outF[(size_t)m * D_ + n] = val + resid[(size_t)m * D_ + n];
      else      outH[(size_t)m * D_ + n] = (_Float16)val;
    }
  }
}

// ---------------------------------------------------------------------------
// Flash attention.  Grid: (B*H, Q/64).  128 threads / 4 waves, each wave a
// 16-row Q tile.  Double-buffered 32-wide S tiles: K staged async, V staged
// one tile ahead through registers with an LDS transpose.  Per S-tile:
// 2 score WMMAs, mask+scale, online softmax, P via per-wave LDS, 2 P*V WMMAs.
// ---------------------------------------------------------------------------
__global__ __launch_bounds__(128)
void attn_kernel(const _Float16* __restrict__ qF, const _Float16* __restrict__ kF,
                 const _Float16* __restrict__ vF, const unsigned char* __restrict__ mask,
                 _Float16* __restrict__ outF) {
  __shared__ __align__(16) _Float16 lK[2][32 * 32];    // [s'][hd]
  __shared__ __align__(16) _Float16 lVT[2][32 * 32];   // [hd][s'] (transposed)
  __shared__ __align__(16) _Float16 lP[4][16 * 32];    // per-wave P tile

  const int t    = threadIdx.x;
  const int wave = t >> 5, lane = t & 31;
  const int b    = blockIdx.x / H_, head = blockIdx.x % H_;
  const int qbase = blockIdx.y * 64 + wave * 16;
  const int r = lane & 15, hg = lane >> 4;
  const float scale = 0.17677669529663687f;            // 1/sqrt(HD)

  // Q fragment for this wave's 16 rows (A layout, K=HD=32), loaded once.
  F16Frag qa;
  {
    const _Float16* qp = qF + ((size_t)(b * Q_) + qbase + r) * D_ + head * HD_;
    qa.q[0] = *reinterpret_cast<const uint4*>(qp + hg * 8);
    qa.q[1] = *reinterpret_cast<const uint4*>(qp + 16 + hg * 8);
  }

  const v8f vzero = {0.f, 0.f, 0.f, 0.f, 0.f, 0.f, 0.f, 0.f};
  v8f O0 = vzero, O1 = vzero;
  float rm[8], rl[8];
#pragma unroll
  for (int j = 0; j < 8; ++j) { rm[j] = -__builtin_inff(); rl[j] = 0.f; }

  // tile loader: thread -> (S-row, 16B segment)
  const int srow = t >> 2, seg = t & 3;
  const _Float16* gK = kF + ((size_t)(b * S_) + srow) * D_ + head * HD_ + seg * 8;
  const _Float16* gV = vF + ((size_t)(b * S_) + srow) * D_ + head * HD_ + seg * 8;
  const size_t stepS = (size_t)32 * D_;
  const int lko = srow * 32 + seg * 8;

  // ---- prologue: stage S-tile 0
#if ASYNC_LDS
  STAGE16(gK, &lK[0][lko]);
#else
  *reinterpret_cast<uint4*>(&lK[0][lko]) = *reinterpret_cast<const uint4*>(gK);
#endif
  {
    F16x8 v0; v0.u = *reinterpret_cast<const uint4*>(gV);
#pragma unroll
    for (int e = 0; e < 8; ++e) lVT[0][(seg * 8 + e) * 32 + srow] = v0.h[e];
  }
#if ASYNC_LDS
  async_wait_all();
#endif
  __syncthreads();

  for (int st = 0; st < S_ / 32; ++st) {
    const int sbase = st * 32;
    const int cur = st & 1, nxt = cur ^ 1;
    const bool more = (st + 1 < S_ / 32);

    // ---- stage next tile: K async, V into registers (transpose-stored later)
    F16x8 vn;
#if ASYNC_LDS
    if (more) STAGE16(gK + (st + 1) * stepS, &lK[nxt][lko]);
#else
    uint4 kreg;
    if (more) kreg = *reinterpret_cast<const uint4*>(gK + (st + 1) * stepS);
#endif
    if (more) vn.u = *reinterpret_cast<const uint4*>(gV + (st + 1) * stepS);

    // ---- scores = Q * K^T on current buffer
    F16Frag kb0, kb1;
    {
      const _Float16* k0 = &lK[cur][r * 32 + hg * 8];
      kb0.q[0] = *reinterpret_cast<const uint4*>(k0);
      kb0.q[1] = *reinterpret_cast<const uint4*>(k0 + 16);
      const _Float16* k1 = &lK[cur][(16 + r) * 32 + hg * 8];
      kb1.q[0] = *reinterpret_cast<const uint4*>(k1);
      kb1.q[1] = *reinterpret_cast<const uint4*>(k1 + 16);
    }
    v8f c0 = __builtin_amdgcn_wmma_f32_16x16x32_f16(
        false, qa.v, false, kb0.v, (short)0, vzero, false, false);
    v8f c1 = __builtin_amdgcn_wmma_f32_16x16x32_f16(
        false, qa.v, false, kb1.v, (short)0, vzero, false, false);

    // ---- mask + scale + online softmax
#pragma unroll
    for (int j = 0; j < 8; ++j) {
      const int qrow = qbase + j + 8 * hg;
      const size_t mi = ((size_t)b * Q_ + qrow) * (size_t)S_ + sbase;
      float s0 = mask[mi + r]      ? c0[j] * scale : NEGV;
      float s1 = mask[mi + 16 + r] ? c1[j] * scale : NEGV;
      float tmax = fmaxf(s0, s1);
#pragma unroll
      for (int mk = 1; mk < 16; mk <<= 1) tmax = fmaxf(tmax, __shfl_xor(tmax, mk, 32));
      float newm  = fmaxf(rm[j], tmax);
      float alpha = __expf(rm[j] - newm);
      float p0 = __expf(s0 - newm);
      float p1 = __expf(s1 - newm);
      float ps = p0 + p1;
#pragma unroll
      for (int mk = 1; mk < 16; mk <<= 1) ps += __shfl_xor(ps, mk, 32);
      rl[j] = rl[j] * alpha + ps;
      rm[j] = newm;
      O0[j] *= alpha;
      O1[j] *= alpha;
      const int prow = j + 8 * hg;
      lP[wave][prow * 32 + r]      = (_Float16)p0;
      lP[wave][prow * 32 + 16 + r] = (_Float16)p1;
    }

    // ---- reload P as A fragment (DS ops are in-order within a wave)
    F16Frag pa;
    {
      const _Float16* pb = &lP[wave][r * 32 + hg * 8];
      pa.q[0] = *reinterpret_cast<const uint4*>(pb);
      pa.q[1] = *reinterpret_cast<const uint4*>(pb + 16);
    }
    // ---- V fragments from transposed tile: lane col n = hd, K = s' in table order
    F16Frag vb0, vb1;
    {
      const _Float16* v0 = &lVT[cur][r * 32 + hg * 8];
      vb0.q[0] = *reinterpret_cast<const uint4*>(v0);
      vb0.q[1] = *reinterpret_cast<const uint4*>(v0 + 16);
      const _Float16* v1 = &lVT[cur][(16 + r) * 32 + hg * 8];
      vb1.q[0] = *reinterpret_cast<const uint4*>(v1);
      vb1.q[1] = *reinterpret_cast<const uint4*>(v1 + 16);
    }
    O0 = __builtin_amdgcn_wmma_f32_16x16x32_f16(
        false, pa.v, false, vb0.v, (short)0, O0, false, false);
    O1 = __builtin_amdgcn_wmma_f32_16x16x32_f16(
        false, pa.v, false, vb1.v, (short)0, O1, false, false);

    // ---- finish staging next tile
    if (more) {
#if !ASYNC_LDS
      *reinterpret_cast<uint4*>(&lK[nxt][lko]) = kreg;
#endif
#pragma unroll
      for (int e = 0; e < 8; ++e) lVT[nxt][(seg * 8 + e) * 32 + srow] = vn.h[e];
    }
#if ASYNC_LDS
    async_wait_all();
#endif
    __syncthreads();
  }

  // ---- normalize and write f16 [B,Q,D] (head folded into columns)
#pragma unroll
  for (int j = 0; j < 8; ++j) {
    const float inv = 1.0f / rl[j];
    const int qrow = qbase + j + 8 * hg;
    _Float16* op = outF + ((size_t)(b * Q_) + qrow) * D_ + head * HD_;
    op[r]      = (_Float16)(O0[j] * inv);
    op[16 + r] = (_Float16)(O1[j] * inv);
  }
}

// ---------------------------------------------------------------------------
// Host-side launch.  Input order: query, key, value, attention_mask,
// wq, bq, wk, bk, wv, bv, wo, bo, ln_g, ln_b.
// ---------------------------------------------------------------------------
extern "C" void kernel_launch(void* const* d_in, const int* in_sizes, int n_in,
                              void* d_out, int out_size, void* d_ws, size_t ws_size,
                              hipStream_t stream) {
  const float* query = (const float*)d_in[0];
  const float* key   = (const float*)d_in[1];
  const float* value = (const float*)d_in[2];
  const unsigned char* mask = (const unsigned char*)d_in[3];  // jnp bool (1 byte)
  const float* wq = (const float*)d_in[4];
  const float* bq = (const float*)d_in[5];
  const float* wk = (const float*)d_in[6];
  const float* bk = (const float*)d_in[7];
  const float* wv = (const float*)d_in[8];
  const float* bv = (const float*)d_in[9];
  const float* wo = (const float*)d_in[10];
  const float* bo = (const float*)d_in[11];
  const float* lng = (const float*)d_in[12];
  const float* lnb = (const float*)d_in[13];
  float* out = (float*)d_out;

  const size_t BQD = (size_t)B_ * Q_ * D_;
  const size_t BSD = (size_t)B_ * S_ * D_;
  const size_t DD  = (size_t)D_ * D_;

  char* ws = (char*)d_ws;
  _Float16* xln   = (_Float16*)ws;  ws += BQD * 2;
  _Float16* keyH  = (_Float16*)ws;  ws += BSD * 2;
  _Float16* valH  = (_Float16*)ws;  ws += BSD * 2;
  _Float16* wH    = (_Float16*)ws;  ws += 4 * DD * 2;
  _Float16* qH    = (_Float16*)ws;  ws += BQD * 2;
  _Float16* kH    = (_Float16*)ws;  ws += BSD * 2;
  _Float16* vH    = (_Float16*)ws;  ws += BSD * 2;
  _Float16* attnH = (_Float16*)ws;  ws += BQD * 2;

  // 1) LayerNorm(query) -> f16
  ln_kernel<<<(B_ * Q_) / 8, 256, 0, stream>>>(query, lng, lnb, xln);

  // 2) f32 -> f16 casts
  cast_kernel<<<(int)((BSD + 255) / 256), 256, 0, stream>>>(key,   keyH, (int)BSD);
  cast_kernel<<<(int)((BSD + 255) / 256), 256, 0, stream>>>(value, valH, (int)BSD);
  cast_kernel<<<(int)((DD + 255) / 256), 256, 0, stream>>>(wq, wH + 0 * DD, (int)DD);
  cast_kernel<<<(int)((DD + 255) / 256), 256, 0, stream>>>(wk, wH + 1 * DD, (int)DD);
  cast_kernel<<<(int)((DD + 255) / 256), 256, 0, stream>>>(wv, wH + 2 * DD, (int)DD);
  cast_kernel<<<(int)((DD + 255) / 256), 256, 0, stream>>>(wo, wH + 3 * DD, (int)DD);

  // 3) Projections (WMMA GEMMs)
  dim3 gq((B_ * Q_) / 64, D_ / 64);
  dim3 gs((B_ * S_) / 64, D_ / 64);
  gemm256_kernel<<<gq, 128, 0, stream>>>(xln,  wH + 0 * DD, bq, qH, nullptr, nullptr);
  gemm256_kernel<<<gs, 128, 0, stream>>>(keyH, wH + 1 * DD, bk, kH, nullptr, nullptr);
  gemm256_kernel<<<gs, 128, 0, stream>>>(valH, wH + 2 * DD, bv, vH, nullptr, nullptr);

  // 4) Flash attention
  dim3 ga(B_ * H_, Q_ / 64);
  attn_kernel<<<ga, 128, 0, stream>>>(qH, kH, vH, mask, attnH);

  // 5) Output projection + bias + residual (f32 out)
  gemm256_kernel<<<gq, 128, 0, stream>>>(attnH, wH + 3 * DD, bo, nullptr, out, query);
}
